// SentenceLevelCnnLstmWithAttention_53420803228387
// MI455X (gfx1250) — matrix operand order repaired and checked by reading, not compiled, not run on hardware
//
#include <hip/hip_runtime.h>
#include <hip/hip_bf16.h>

// ---------------------------------------------------------------------------
// SentenceLevelCnnLstmWithAttention on gfx1250 (MI455X, wave32, WMMA).
// conv-as-GEMM via "sliding flat view" im2col in LDS; all matmuls on
// v_wmma_f32_16x16x32_f16 with f32 accumulation. Waves own disjoint N-tiles
// so each weight fragment is fetched once per block (B-traffic floor).
// ---------------------------------------------------------------------------

typedef __attribute__((ext_vector_type(16))) _Float16 v16h;
typedef __attribute__((ext_vector_type(8)))  _Float16 v8h;
typedef __attribute__((ext_vector_type(8)))  float    v8f;

#define B_ 32
#define E_ 64
#define S_ 50
#define D_ 300
#define F_ 100
#define H_ 100

// workspace byte offsets (all 256B aligned)
#define OFF_ESSH   ((size_t)0)          // f16 essays           [B*E*S*D]
#define OFF_BTCONV ((size_t)12288000)   // f16 conv weights^T   [112][1600]
#define OFF_AT1    ((size_t)12646400)   // f16 att1_mat^T       [112][128]
#define OFF_WT     ((size_t)12675072)   // f16 lstm_W^T         [400][224]
#define OFF_SENTH  ((size_t)12854272)   // f16 sentence reprs   [B*E][112]
#define OFF_LSTMO  ((size_t)13313024)   // f32 lstm hidden seq  [B*E][100]

// ---------------------------------------------------------------------------
// WMMA helpers (CDNA5 16x16x32 f16 -> f32)
// ---------------------------------------------------------------------------
__device__ __forceinline__ v8f wmma_16x16x32(v16h a, v16h b, v8f c) {
  return __builtin_amdgcn_wmma_f32_16x16x32_f16(false, a, false, b,
                                                (short)0, c, false, false);
}

// A fragment (16x32, row-major source, stride in f16 elements).
// Layout used here (same k order as B): lane m=l&15; lane-group g=l>>4; elem i -> k = 16g + i (contiguous)
__device__ __forceinline__ v16h load_a_frag(const _Float16* p, int stride) {
  const int lane = threadIdx.x & 31;
  return *(const v16h*)(p + (lane & 15) * stride + (lane >> 4) * 16);
}

// B fragment from an N-major (transposed, [n][k]) matrix.
// Layout used here (same k order as A): lane n=l&15; elem i -> k = 16*(l>>4) + i  (contiguous)
__device__ __forceinline__ v16h load_bt_frag(const _Float16* p, int stride) {
  const int lane = threadIdx.x & 31;
  return *(const v16h*)(p + (lane & 15) * stride + (lane >> 4) * 16);
}

__device__ __forceinline__ float sigmoidf_(float x) {
  return 1.0f / (1.0f + __expf(-x));
}

// ---------------------------------------------------------------------------
// Prep kernels: f32 -> f16 conversions + weight transposes with K/N padding
// ---------------------------------------------------------------------------
__global__ void prep_f32_to_f16(const float* __restrict__ x,
                                _Float16* __restrict__ y, int n) {
  for (int i = blockIdx.x * blockDim.x + threadIdx.x; i < n;
       i += gridDim.x * blockDim.x)
    y[i] = (_Float16)x[i];
}

// conv_k (1,5,300,100) HWIO -> Bt[112][1600], k = w*320 + d (d padded to 320)
__global__ void prep_conv_bt(const float* __restrict__ k,
                             _Float16* __restrict__ bt) {
  for (int idx = blockIdx.x * blockDim.x + threadIdx.x; idx < 112 * 1600;
       idx += gridDim.x * blockDim.x) {
    int n = idx / 1600, kk = idx - n * 1600;
    int w = kk / 320, d = kk - w * 320;
    float v = (n < F_ && d < D_) ? k[(w * D_ + d) * F_ + n] : 0.f;
    bt[idx] = (_Float16)v;
  }
}

// att1_mat (F,F) -> Bt[112][128]
__global__ void prep_att1_bt(const float* __restrict__ m,
                             _Float16* __restrict__ bt) {
  for (int idx = blockIdx.x * blockDim.x + threadIdx.x; idx < 112 * 128;
       idx += gridDim.x * blockDim.x) {
    int n = idx / 128, kk = idx - n * 128;
    float v = (n < F_ && kk < F_) ? m[kk * F_ + n] : 0.f;
    bt[idx] = (_Float16)v;
  }
}

// lstm_W (200,400) -> Bt[400][224]; x rows at k<100, h rows at k in [112,212)
__global__ void prep_lstm_bt(const float* __restrict__ w,
                             _Float16* __restrict__ bt) {
  for (int idx = blockIdx.x * blockDim.x + threadIdx.x; idx < 400 * 224;
       idx += gridDim.x * blockDim.x) {
    int n = idx / 224, kk = idx - n * 224;
    int kp = (kk < 100) ? kk : ((kk >= 112 && kk < 212) ? 100 + (kk - 112) : -1);
    float v = (kp >= 0) ? w[kp * 400 + n] : 0.f;
    bt[idx] = (_Float16)v;
  }
}

// ---------------------------------------------------------------------------
// Kernel 1: conv (im2col WMMA GEMM) + word attention pooling.
// One block per (b,e); 128 threads = 4 waves. Wave w owns N-tiles {w, w+4}
// (wave 3 owns only tile 3) and computes ALL 4 M-tiles, so every weight
// fragment is loaded from global exactly once per block.
// ---------------------------------------------------------------------------
#define SROWS 70
#define SSTR  320
#define CSTR  128

__global__ __launch_bounds__(128) void conv_att1_kernel(
    const _Float16* __restrict__ essH, const _Float16* __restrict__ btConv,
    const _Float16* __restrict__ btAtt1, const float* __restrict__ convB,
    const float* __restrict__ a1bias, const float* __restrict__ a1vec,
    _Float16* __restrict__ sentH) {
  __shared__ __align__(32) _Float16 sA[SROWS * SSTR];   // padded sentence (f16)
  __shared__ __align__(32) _Float16 sConv[64 * CSTR];   // relu(conv1) (f16)
  __shared__ float sScore[64];
  __shared__ float sInv[1];

  const int tid  = threadIdx.x;
  const int be   = blockIdx.x;
  const int wave = tid >> 5;
  const int lane = tid & 31;
  const int g    = lane >> 4;
  const int nt0  = wave;               // first owned N-tile
  const bool has2 = (wave < 3);        // second owned N-tile = wave + 4

  // zero-init padded buffers + scores, then stage sentence (+2 row pad, 8B cps)
  for (int i = tid; i < SROWS * SSTR; i += 128) sA[i] = (_Float16)0.f;
  for (int i = tid; i < 64 * CSTR;    i += 128) sConv[i] = (_Float16)0.f;
  if (tid < 64) sScore[tid] = 0.f;
  __syncthreads();
  const _Float16* gx = essH + (size_t)be * (S_ * D_);
  for (int i = tid; i < (S_ * D_) / 4; i += 128) {   // 3750 x 8-byte chunks
    int s = i / 75, r = i - s * 75;                  // 75 chunks per row
    *(uint2*)(&sA[(s + 2) * SSTR + r * 4]) = *(const uint2*)(gx + i * 4);
  }
  __syncthreads();

  // ---- GEMM1: conv1 = im2col(sent) @ Wc   (M=64, N=112, K=1600) ----
  v8f acc[4][2];
#pragma unroll
  for (int m = 0; m < 4; ++m)
#pragma unroll
    for (int j = 0; j < 2; ++j)
#pragma unroll
      for (int q = 0; q < 8; ++q) acc[m][j][q] = 0.f;

  for (int kt = 0; kt < 50; ++kt) {
    v16h afr[4];
#pragma unroll
    for (int m = 0; m < 4; ++m)      // im2col A row m = sA + m*SSTR (flat view)
      afr[m] = load_a_frag(sA + m * 16 * SSTR + kt * 32, SSTR);
    v16h b0 = load_bt_frag(btConv + nt0 * 16 * 1600 + kt * 32, 1600);
#pragma unroll
    for (int m = 0; m < 4; ++m) acc[m][0] = wmma_16x16x32(afr[m], b0, acc[m][0]);
    if (has2) {
      v16h b1 = load_bt_frag(btConv + (nt0 + 4) * 16 * 1600 + kt * 32, 1600);
#pragma unroll
      for (int m = 0; m < 4; ++m) acc[m][1] = wmma_16x16x32(afr[m], b1, acc[m][1]);
    }
  }
  // bias + relu -> f16 LDS (padded cols stay zero: Bt cols >=100 are zero)
#pragma unroll
  for (int j = 0; j < 2; ++j) {
    if (j == 0 || has2) {
      int n = (nt0 + 4 * j) * 16 + (lane & 15);
      float bn = (n < F_) ? convB[n] : 0.f;
#pragma unroll
      for (int m = 0; m < 4; ++m)
#pragma unroll
        for (int r = 0; r < 8; ++r) {
          int mm = m * 16 + r + 8 * g;
          float v = acc[m][j][r] + bn;
          sConv[mm * CSTR + n] = (_Float16)(v > 0.f ? v : 0.f);
        }
    }
  }
  __syncthreads();

  // ---- GEMM2: u = conv1 @ att1_mat   (M=64, N=112, K=128) ----
  v8f acc2[4][2];
#pragma unroll
  for (int m = 0; m < 4; ++m)
#pragma unroll
    for (int j = 0; j < 2; ++j)
#pragma unroll
      for (int q = 0; q < 8; ++q) acc2[m][j][q] = 0.f;

  for (int kt = 0; kt < 4; ++kt) {
    v16h afr[4];
#pragma unroll
    for (int m = 0; m < 4; ++m)
      afr[m] = load_a_frag(sConv + m * 16 * CSTR + kt * 32, CSTR);
    v16h b0 = load_bt_frag(btAtt1 + nt0 * 16 * 128 + kt * 32, 128);
#pragma unroll
    for (int m = 0; m < 4; ++m) acc2[m][0] = wmma_16x16x32(afr[m], b0, acc2[m][0]);
    if (has2) {
      v16h b1 = load_bt_frag(btAtt1 + (nt0 + 4) * 16 * 128 + kt * 32, 128);
#pragma unroll
      for (int m = 0; m < 4; ++m) acc2[m][1] = wmma_16x16x32(afr[m], b1, acc2[m][1]);
    }
  }

  // partial scores over this wave's N-columns, half-wave reduce, LDS atomicAdd
  float pr[4][8];
#pragma unroll
  for (int m = 0; m < 4; ++m)
#pragma unroll
    for (int r = 0; r < 8; ++r) pr[m][r] = 0.f;
#pragma unroll
  for (int j = 0; j < 2; ++j) {
    if (j == 0 || has2) {
      int n = (nt0 + 4 * j) * 16 + (lane & 15);
      float bn = (n < F_) ? a1bias[n] : 0.f;
      float vn = (n < F_) ? a1vec[n] : 0.f;
#pragma unroll
      for (int m = 0; m < 4; ++m)
#pragma unroll
        for (int r = 0; r < 8; ++r)
          pr[m][r] += tanhf(acc2[m][j][r] + bn) * vn;
    }
  }
#pragma unroll
  for (int m = 0; m < 4; ++m)
#pragma unroll
    for (int r = 0; r < 8; ++r) {
#pragma unroll
      for (int off = 1; off < 16; off <<= 1)
        pr[m][r] += __shfl_xor(pr[m][r], off, 32);
    }
  if ((lane & 15) == 0) {
#pragma unroll
    for (int m = 0; m < 4; ++m)
#pragma unroll
      for (int r = 0; r < 8; ++r)
        atomicAdd(&sScore[m * 16 + r + 8 * g], pr[m][r]);
  }
  __syncthreads();

  // softmax over words (only s<50 valid)
  if (tid == 0) {
    float mx = -1e30f;
    for (int s = 0; s < S_; ++s) mx = fmaxf(mx, sScore[s]);
    float sum = 0.f;
    for (int s = 0; s < S_; ++s) {
      float e = __expf(sScore[s] - mx);
      sScore[s] = e;
      sum += e;
    }
    sInv[0] = 1.f / sum;
  }
  __syncthreads();

  // sentence repr = sum_s w_s * conv1[s,:]
  if (tid < 112) {
    float inv = sInv[0], a = 0.f;
    for (int s = 0; s < S_; ++s)
      a += sScore[s] * inv * (float)sConv[s * CSTR + tid];
    sentH[(size_t)be * 112 + tid] = (_Float16)((tid < F_) ? a : 0.f);
  }
}

// ---------------------------------------------------------------------------
// Kernel 2: LSTM over E=64 sentences. One block, 256 threads (8 waves).
// Per step: z = [x;h] @ W + b via WMMA (M=32, N=400, K=224), gate math in regs.
// ---------------------------------------------------------------------------
__global__ __launch_bounds__(256) void lstm_kernel(
    const _Float16* __restrict__ sentH, const _Float16* __restrict__ wt,
    const float* __restrict__ bias, float* __restrict__ hout) {
  __shared__ __align__(32) _Float16 sAh[32 * 224];  // [b][k]: x | pad | h | pad
  __shared__ float sZ[32 * 400];

  const int tid = threadIdx.x, wave = tid >> 5, lane = tid & 31;
  const int g = lane >> 4;

  float creg[13];
#pragma unroll
  for (int j = 0; j < 13; ++j) creg[j] = 0.f;
  for (int i = tid; i < 32 * 224; i += 256) sAh[i] = (_Float16)0.f;
  __syncthreads();

  for (int t = 0; t < E_; ++t) {
    // load x_t (f16) into A cols [0,100)
    for (int i = tid; i < 32 * F_; i += 256) {
      int b = i / F_, u = i - b * F_;
      sAh[b * 224 + u] = sentH[(size_t)(b * E_ + t) * 112 + u];
    }
    __syncthreads();

    // z = A @ W^T-layout  (25 N-tiles over 8 waves)
    for (int nt = wave; nt < 25; nt += 8) {
      v8f a0, a1;
#pragma unroll
      for (int j = 0; j < 8; ++j) { a0[j] = 0.f; a1[j] = 0.f; }
      for (int kt = 0; kt < 7; ++kt) {
        v16h b  = load_bt_frag(wt + nt * 16 * 224 + kt * 32, 224);
        v16h x0 = load_a_frag(sAh + kt * 32, 224);
        v16h x1 = load_a_frag(sAh + 16 * 224 + kt * 32, 224);
        a0 = wmma_16x16x32(x0, b, a0);
        a1 = wmma_16x16x32(x1, b, a1);
      }
      int n = nt * 16 + (lane & 15);
      float bn = bias[n];
#pragma unroll
      for (int r = 0; r < 8; ++r) {
        sZ[(r + 8 * g) * 400 + n]      = a0[r] + bn;
        sZ[(16 + r + 8 * g) * 400 + n] = a1[r] + bn;
      }
    }
    __syncthreads();

    // gate math: c in registers; write h back (f16) + to global (f32)
#pragma unroll
    for (int j = 0; j < 13; ++j) {
      int p = tid + 256 * j;
      if (p < 32 * H_) {
        int b = p / H_, u = p - b * H_;
        const float* zr = sZ + b * 400;
        float zi = zr[u], zj = zr[100 + u], zf = zr[200 + u], zo = zr[300 + u];
        float c = creg[j];
        c = c * sigmoidf_(zf + 1.0f) + sigmoidf_(zi) * tanhf(zj);
        float h = sigmoidf_(zo) * tanhf(c);
        creg[j] = c;
        sAh[b * 224 + 112 + u] = (_Float16)h;
        hout[(size_t)(b * E_ + t) * H_ + u] = h;
      }
    }
    __syncthreads();
  }
}

// ---------------------------------------------------------------------------
// Kernel 3: sentence attention + dense sigmoid. One block per batch row.
// ---------------------------------------------------------------------------
__global__ __launch_bounds__(128) void att2_dense_kernel(
    const float* __restrict__ lstmo, const float* __restrict__ a2mat,
    const float* __restrict__ a2bias, const float* __restrict__ a2vec,
    const float* __restrict__ dw, const float* __restrict__ db,
    float* __restrict__ out) {
  __shared__ float sU[128];
  __shared__ float sSc[64];
  const int tid = threadIdx.x;
  const int b = blockIdx.x;

  for (int e = 0; e < E_; ++e) {
    float u = 0.f;
    if (tid < H_) {
      const float* hr = lstmo + (size_t)(b * E_ + e) * H_;
      float s = 0.f;
      for (int h = 0; h < H_; ++h) s += hr[h] * a2mat[h * H_ + tid];
      u = tanhf(s + a2bias[tid]) * a2vec[tid];
    }
    sU[tid] = u;
    __syncthreads();
    for (int off = 64; off > 0; off >>= 1) {
      if (tid < off) sU[tid] += sU[tid + off];
      __syncthreads();
    }
    if (tid == 0) sSc[e] = sU[0];
    __syncthreads();
  }
  if (tid == 0) {
    float mx = -1e30f;
    for (int e = 0; e < E_; ++e) mx = fmaxf(mx, sSc[e]);
    float sum = 0.f;
    for (int e = 0; e < E_; ++e) { sSc[e] = __expf(sSc[e] - mx); sum += sSc[e]; }
    float inv = 1.f / sum;
    for (int e = 0; e < E_; ++e) sSc[e] *= inv;
  }
  __syncthreads();
  float v = 0.f;
  if (tid < H_) {
    float rep = 0.f;
    for (int e = 0; e < E_; ++e)
      rep += sSc[e] * lstmo[(size_t)(b * E_ + e) * H_ + tid];
    v = rep * dw[tid];
  }
  sU[tid] = v;
  __syncthreads();
  for (int off = 64; off > 0; off >>= 1) {
    if (tid < off) sU[tid] += sU[tid + off];
    __syncthreads();
  }
  if (tid == 0) out[b] = 1.0f / (1.0f + __expf(-(sU[0] + db[0])));
}

// ---------------------------------------------------------------------------
extern "C" void kernel_launch(void* const* d_in, const int* in_sizes, int n_in,
                              void* d_out, int out_size, void* d_ws,
                              size_t ws_size, hipStream_t stream) {
  (void)in_sizes; (void)n_in; (void)out_size; (void)ws_size;
  const float* essays    = (const float*)d_in[0];
  const float* conv_k    = (const float*)d_in[1];
  const float* conv_b    = (const float*)d_in[2];
  const float* att1_mat  = (const float*)d_in[3];
  const float* att1_bias = (const float*)d_in[4];
  const float* att1_vec  = (const float*)d_in[5];
  const float* lstm_W    = (const float*)d_in[6];
  const float* lstm_b    = (const float*)d_in[7];
  const float* att2_mat  = (const float*)d_in[8];
  const float* att2_bias = (const float*)d_in[9];
  const float* att2_vec  = (const float*)d_in[10];
  const float* dense_w   = (const float*)d_in[11];
  const float* dense_b   = (const float*)d_in[12];

  char* ws = (char*)d_ws;
  _Float16* essH   = (_Float16*)(ws + OFF_ESSH);
  _Float16* btConv = (_Float16*)(ws + OFF_BTCONV);
  _Float16* btAtt1 = (_Float16*)(ws + OFF_AT1);
  _Float16* btW    = (_Float16*)(ws + OFF_WT);
  _Float16* sentH  = (_Float16*)(ws + OFF_SENTH);
  float*    lstmo  = (float*)(ws + OFF_LSTMO);

  prep_f32_to_f16<<<2048, 256, 0, stream>>>(essays, essH, B_ * E_ * S_ * D_);
  prep_conv_bt<<<350, 256, 0, stream>>>(conv_k, btConv);
  prep_att1_bt<<<56, 256, 0, stream>>>(att1_mat, btAtt1);
  prep_lstm_bt<<<175, 256, 0, stream>>>(lstm_W, btW);

  conv_att1_kernel<<<B_ * E_, 128, 0, stream>>>(essH, btConv, btAtt1, conv_b,
                                                att1_bias, att1_vec, sentH);
  lstm_kernel<<<1, 256, 0, stream>>>(sentH, btW, lstm_b, lstmo);
  att2_dense_kernel<<<B_, 128, 0, stream>>>(lstmo, att2_mat, att2_bias,
                                            att2_vec, dense_w, dense_b,
                                            (float*)d_out);
}
